// MixtralSparseMoeBlock_24446953849472
// MI455X (gfx1250) — matrix operand order repaired and checked by reading, not compiled, not run on hardware
//
#include <hip/hip_runtime.h>
#include <cstdint>
#include <cstddef>

// ---------------------------------------------------------------------------
// Mixtral-style sparse MoE block for gfx1250 (MI455X), bf16 WMMA path.
//   hidden_states [2,2048,1024] f32, router [1024,8] f32,
//   w_gate/w_inter [8,1024,3584] f32, w_out [8,3584,1024] f32
//   out = concat( moe_out [4096,1024] f32 , router_logits [4096,8] f32 )
// M-tile = 32 tokens: every B fragment feeds two WMMAs (M-subtiles), halving
// the global-load + cvt issue per FLOP and halving L2 weight re-reads.
// ---------------------------------------------------------------------------

typedef __attribute__((ext_vector_type(16))) __bf16 v16bf;
typedef __attribute__((ext_vector_type(8)))  float  v8f;

namespace {
constexpr int kH = 1024;   // hidden
constexpr int kI = 3584;   // intermediate
constexpr int kE = 8;      // experts
constexpr int kT = 4096;   // tokens (2*2048)
constexpr int kTileM = 32; // tokens per FFN block (2 x 16-row WMMA subtiles)
constexpr int kMaxTiles = kT / kTileM;   // 128 (worst case: all tokens -> one expert)

// LDS layout (dynamic shared memory, ~289 KB; WGP has 320 KB)
constexpr int kXS = kH + 8;   // padded bf16 row stride for x tile (bank spread)
constexpr int kHS = kI + 8;   // padded bf16 row stride for h tile
constexpr size_t kLdsX  = (size_t)kTileM * kXS * sizeof(__bf16);   // 66048
constexpr size_t kLdsH  = (size_t)kTileM * kHS * sizeof(__bf16);   // 229888
constexpr size_t kLdsW  = kTileM * sizeof(float);                  // 128
constexpr size_t kLdsTk = kTileM * sizeof(int);                    // 128
constexpr size_t kLdsBytes = kLdsX + kLdsH + kLdsW + kLdsTk;       // 296192
} // namespace

union AFrag { uint4 q[2]; v16bf v; };

// A-fragment (16x32 bf16, K-major rows in LDS). ISA layout: lanes 0-15 hold
// M=lane, K 0-7 in elems 0-7 and K 16-23 in elems 8-15; lanes 16-31 hold
// M=lane-16, K 8-15 / 24-31.  Both halves are K-contiguous -> two 16B loads.
static __device__ inline v16bf load_a_bf16(const __bf16* row_base, int k0, int lane) {
    const int kb = (lane & 16) ? 8 : 0;
    AFrag f;
    f.q[0] = *reinterpret_cast<const uint4*>(row_base + k0 + kb);
    f.q[1] = *reinterpret_cast<const uint4*>(row_base + k0 + kb + 16);
    return f.v;
}

// B-fragment (32x16) gathered from a row-major fp32 weight matrix W[ldw cols]:
// lane n = lane&15, kb = 0 (lanes 0-15) or 8 (lanes 16-31),
// element j -> K = k0 + kb + j (j<8) or k0 + kb + 8 + j (j>=8).  fp32 -> bf16 conversion in flight.
static __device__ inline v16bf load_b_f32(const float* __restrict__ W, int ldw,
                                          int k0, int col, int lane) {
    const int kb = (lane & 16) ? 8 : 0;
    const float* p = W + (size_t)(k0 + kb) * ldw + col;
    v16bf b;
#pragma unroll
    for (int j = 0; j < 16; ++j)
        b[j] = (__bf16)p[(size_t)(j < 8 ? j : j + 8) * ldw];
    return b;
}

static __device__ inline v8f wmma_bf16(v16bf a, v16bf b, v8f c) {
    return __builtin_amdgcn_wmma_f32_16x16x32_bf16(
        /*neg_a=*/false, a, /*neg_b=*/false, b,
        /*c_mod=*/(short)0, c, /*reuse_a=*/false, /*reuse_b=*/false);
}

// ---------------------------------------------------------------------------
// Kernel 1: zero the accumulated output region + expert counters.
// ---------------------------------------------------------------------------
__global__ __launch_bounds__(256) void moe_init(float* __restrict__ out,
                                                int* __restrict__ cnt) {
    size_t i = (size_t)blockIdx.x * blockDim.x + threadIdx.x;
    float4* o4 = reinterpret_cast<float4*>(out);
    if (i < (size_t)kT * kH / 4) o4[i] = make_float4(0.f, 0.f, 0.f, 0.f);
    if (blockIdx.x == 0 && threadIdx.x < kE) cnt[threadIdx.x] = 0;
}

// ---------------------------------------------------------------------------
// Kernel 2: router.  One wave (32 lanes) per token.
//   logits = x @ rk ; normalized top-2 ; append (token, weight) per expert.
// ---------------------------------------------------------------------------
__global__ __launch_bounds__(128) void moe_router(const float* __restrict__ x,
                                                  const float* __restrict__ rk,
                                                  float* __restrict__ logits,
                                                  int* __restrict__ cnt,
                                                  int* __restrict__ tok,
                                                  float* __restrict__ twv) {
    const int lane = threadIdx.x & 31;
    const int t    = blockIdx.x * 4 + (threadIdx.x >> 5);
    if (t >= kT) return;

    float a[kE];
#pragma unroll
    for (int e = 0; e < kE; ++e) a[e] = 0.f;

    const float* xr = x + (size_t)t * kH;
    for (int h = lane; h < kH; h += 32) {
        const float xv = xr[h];
        const float* r = rk + (size_t)h * kE;
#pragma unroll
        for (int e = 0; e < kE; ++e) a[e] += xv * r[e];
    }
#pragma unroll
    for (int e = 0; e < kE; ++e) {
#pragma unroll
        for (int off = 16; off > 0; off >>= 1)
            a[e] += __shfl_xor(a[e], off, 32);
    }

    if (lane == 0) {
        // raw logits out
#pragma unroll
        for (int e = 0; e < kE; ++e) logits[(size_t)t * kE + e] = a[e];
        // top-2 (first occurrence on ties, matching lax.top_k)
        int i0 = 0; float v0 = a[0];
#pragma unroll
        for (int e = 1; e < kE; ++e) if (a[e] > v0) { v0 = a[e]; i0 = e; }
        int i1 = -1; float v1 = -3.4e38f;
#pragma unroll
        for (int e = 0; e < kE; ++e)
            if (e != i0 && a[e] > v1) { v1 = a[e]; i1 = e; }
        // normalized pair weights: softmax denominator cancels
        const float e1 = __expf(v1 - v0);
        const float w0 = 1.f / (1.f + e1);
        const float w1 = e1 * w0;

        int p0 = atomicAdd(&cnt[i0], 1);
        tok[i0 * kT + p0] = t;  twv[i0 * kT + p0] = w0;
        int p1 = atomicAdd(&cnt[i1], 1);
        tok[i1 * kT + p1] = t;  twv[i1 * kT + p1] = w1;
    }
}

// ---------------------------------------------------------------------------
// Kernel 3: fused expert FFN over one 32-token tile.
//   256 threads = 8 waves.  LDS: x tile (bf16) + full h tile (bf16).
//   Phase A: h = silu(x@Wg) * (x@Wi)   (K = 1024)
//   Phase B: out += w_tok * (h @ Wo)   (K = 3584), atomic f32 adds.
// ---------------------------------------------------------------------------
__global__ __launch_bounds__(256) void moe_ffn(const float* __restrict__ x,
                                               const float* __restrict__ wg,
                                               const float* __restrict__ wi,
                                               const float* __restrict__ wo,
                                               const int*   __restrict__ cnt,
                                               const int*   __restrict__ tok,
                                               const float* __restrict__ twv,
                                               float* __restrict__ out) {
    const int e    = blockIdx.x / kMaxTiles;   // expert-contiguous -> L2 friendly
    const int tile = blockIdx.x % kMaxTiles;
    const int count = cnt[e];
    if (tile * kTileM >= count) return;

    extern __shared__ char smem[];
    __bf16* x_lds = reinterpret_cast<__bf16*>(smem);
    __bf16* h_lds = reinterpret_cast<__bf16*>(smem + kLdsX);
    float*  twgt  = reinterpret_cast<float*>(smem + kLdsX + kLdsH);
    int*    ttok  = reinterpret_cast<int*>(twgt + kTileM);

    const int tid  = threadIdx.x;
    const int lane = tid & 31;
    const int wave = tid >> 5;

    if (tid < kTileM) {
        const int i = tile * kTileM + tid;
        int id = 0; float w = 0.f;
        if (i < count) { id = tok[e * kT + i]; w = twv[e * kT + i]; }
        ttok[tid] = id; twgt[tid] = w;
    }
    __syncthreads();

    // stage x tile -> LDS as bf16
    for (int idx = tid; idx < kTileM * kH; idx += 256) {
        const int r = idx >> 10, k = idx & (kH - 1);
        x_lds[r * kXS + k] = (__bf16)x[(size_t)ttok[r] * kH + k];
    }
    __syncthreads();

    const float* wge = wg + (size_t)e * kH * kI;
    const float* wie = wi + (size_t)e * kH * kI;
    const float* woe = wo + (size_t)e * kI * kH;

    const int ncol  = lane & 15;
    const int mbase = (lane & 16) ? 8 : 0;

    const __bf16* x_row0 = x_lds + (size_t)(lane & 15) * kXS;        // M 0-15
    const __bf16* x_row1 = x_lds + (size_t)(16 + (lane & 15)) * kXS; // M 16-31

    // ---------------- Phase A: gate/up GEMMs + SiLU, h -> LDS --------------
    for (int c0 = 0; c0 < kI; c0 += 128) {
        const int n0 = c0 + wave * 16;          // this wave's 16 columns
        v8f ag0 = {0.f,0.f,0.f,0.f,0.f,0.f,0.f,0.f};
        v8f ag1 = {0.f,0.f,0.f,0.f,0.f,0.f,0.f,0.f};
        v8f au0 = {0.f,0.f,0.f,0.f,0.f,0.f,0.f,0.f};
        v8f au1 = {0.f,0.f,0.f,0.f,0.f,0.f,0.f,0.f};
        for (int k0 = 0; k0 < kH; k0 += 32) {
            const v16bf a0 = load_a_bf16(x_row0, k0, lane);
            const v16bf a1 = load_a_bf16(x_row1, k0, lane);
            const v16bf bg = load_b_f32(wge, kI, k0, n0 + ncol, lane);
            const v16bf bu = load_b_f32(wie, kI, k0, n0 + ncol, lane);
            ag0 = wmma_bf16(a0, bg, ag0);
            ag1 = wmma_bf16(a1, bg, ag1);   // B reuse across M-subtiles
            au0 = wmma_bf16(a0, bu, au0);
            au1 = wmma_bf16(a1, bu, au1);
        }
#pragma unroll
        for (int r = 0; r < 8; ++r) {
            const float g0 = ag0[r], u0 = au0[r];
            const float g1 = ag1[r], u1 = au1[r];
            const float h0 = (g0 / (1.f + __expf(-g0))) * u0;   // silu(g)*u
            const float h1 = (g1 / (1.f + __expf(-g1))) * u1;
            h_lds[(size_t)(r + mbase) * kHS + (n0 + ncol)]      = (__bf16)h0;
            h_lds[(size_t)(16 + r + mbase) * kHS + (n0 + ncol)] = (__bf16)h1;
        }
    }
    __syncthreads();

    // ---------------- Phase B: out = (h @ Wo) * w_tok ----------------------
    const __bf16* h_row0 = h_lds + (size_t)(lane & 15) * kHS;
    const __bf16* h_row1 = h_lds + (size_t)(16 + (lane & 15)) * kHS;
    const int nbase = wave * 128;               // this wave's 128 out columns

    v8f acc[16];                                // [s][m] -> acc[2*s+m]
#pragma unroll
    for (int q = 0; q < 16; ++q)
        acc[q] = (v8f){0.f,0.f,0.f,0.f,0.f,0.f,0.f,0.f};

    for (int k0 = 0; k0 < kI; k0 += 32) {
        const v16bf a0 = load_a_bf16(h_row0, k0, lane);
        const v16bf a1 = load_a_bf16(h_row1, k0, lane);
#pragma unroll
        for (int s = 0; s < 8; ++s) {
            const v16bf b = load_b_f32(woe, kH, k0, nbase + s * 16 + ncol, lane);
            acc[2 * s]     = wmma_bf16(a0, b, acc[2 * s]);      // B reuse
            acc[2 * s + 1] = wmma_bf16(a1, b, acc[2 * s + 1]);
        }
    }

#pragma unroll
    for (int s = 0; s < 8; ++s) {
#pragma unroll
        for (int m = 0; m < 2; ++m) {
#pragma unroll
            for (int r = 0; r < 8; ++r) {
                const int   row = 16 * m + r + mbase;
                const float wv  = twgt[row];
                float* orow = out + (size_t)ttok[row] * kH;
                atomicAdd(&orow[nbase + s * 16 + ncol], acc[2 * s + m][r] * wv);
            }
        }
    }
}

// ---------------------------------------------------------------------------
extern "C" void kernel_launch(void* const* d_in, const int* in_sizes, int n_in,
                              void* d_out, int out_size, void* d_ws, size_t ws_size,
                              hipStream_t stream) {
    (void)in_sizes; (void)n_in; (void)out_size; (void)ws_size;
    const float* x  = (const float*)d_in[0];   // [T, H]
    const float* rk = (const float*)d_in[1];   // [H, E]
    const float* wg = (const float*)d_in[2];   // [E, H, I]
    const float* wi = (const float*)d_in[3];   // [E, H, I]
    const float* wo = (const float*)d_in[4];   // [E, I, H]

    float* out    = (float*)d_out;             // [T, H]
    float* logits = out + (size_t)kT * kH;     // [T, E]

    int*   cnt = (int*)d_ws;                   // [16] (8 used, padded)
    int*   tok = cnt + 16;                     // [E, T]
    float* twv = (float*)(tok + kE * kT);      // [E, T]

    moe_init<<<(kT * kH / 4 + 255) / 256, 256, 0, stream>>>(out, cnt);
    moe_router<<<kT / 4, 128, 0, stream>>>(x, rk, logits, cnt, tok, twv);
    moe_ffn<<<kE * kMaxTiles, 256, (int)kLdsBytes, stream>>>(
        x, wg, wi, wo, cnt, tok, twv, out);
}
